// AnchorKNN_OnlyL_21629455303118
// MI455X (gfx1250) — matrix operand order repaired and checked by reading, not compiled, not run on hardware
//
#include <hip/hip_runtime.h>
#include <hip/hip_bf16.h>

#define BATCH 32768
#define MANC  2048
#define EMB   128
#define KSEL  8
#define TAU   0.3f

typedef __attribute__((ext_vector_type(16))) __bf16 v16bf;
typedef __attribute__((ext_vector_type(8)))  float  v8f;

__device__ __forceinline__ float gelu_exact(float v) {
    return 0.5f * v * (1.0f + erff(v * 0.70710678118654752440f));
}

// ---------------------------------------------------------------------------
// Kernel 0: split W2 (f32 [f][e], row-major) into bf16 hi/lo, packed as dwords
// of (e, e+1) pairs, laid out [ep][f] with an additive (f + ep) & 127 bank swizzle so the fused
// kernel's LDS B-fragment reads are conflict-free across the wave.
// ---------------------------------------------------------------------------
__global__ __launch_bounds__(256) void k_split_w2(const float* __restrict__ W2,
                                                  unsigned* __restrict__ W2h,
                                                  unsigned* __restrict__ W2l) {
    int id = blockIdx.x * 256 + threadIdx.x;   // 0..8191
    int f  = id >> 6;                          // out-feature 0..127
    int ep = id & 63;                          // e-pair 0..63
    float a = W2[f * EMB + 2 * ep + 0];
    float b = W2[f * EMB + 2 * ep + 1];
    __bf16 ah = (__bf16)a;
    __bf16 bh = (__bf16)b;
    __bf16 al = (__bf16)(a - (float)ah);
    __bf16 bl = (__bf16)(b - (float)bh);
    union { __bf16 h[2]; unsigned u; } ph, pl;
    ph.h[0] = ah; ph.h[1] = bh;
    pl.h[0] = al; pl.h[1] = bl;
    int idx = ep * 128 + ((f + ep) & 127);     // bank-swizzled [ep][f]
    W2h[idx] = ph.u;
    W2l[idx] = pl.u;
}

// ---------------------------------------------------------------------------
// Fused kernel: one 256-thread block handles 16 batch rows end-to-end.
// Phase 1 (HBM-bound): each wave does KNN top-8 + softmax weights for 2 rows,
//   writing {x, y, w, d2} to LDS.
// Phase 2 (WMMA-bound): each wave computes 16 GEMM rows (= its 2 batch rows
//   x 8 neighbors) through the bf16-split WMMA GEMM vs W2, GELU, and the
//   softmax-weighted reduction over K.
// Blocks at different phases overlap HBM streaming with matrix math.
// ---------------------------------------------------------------------------
__global__ __launch_bounds__(256) void k_fused(const float* __restrict__ Gl,
                                               const float* __restrict__ ancL,
                                               const float* __restrict__ W1,
                                               const float* __restrict__ b1,
                                               const float* __restrict__ b2,
                                               const unsigned* __restrict__ W2h,
                                               const unsigned* __restrict__ W2l,
                                               float* __restrict__ out) {
    __shared__ unsigned sH[8192];
    __shared__ unsigned sL[8192];
    __shared__ float4   smeta[128];            // 16 batch rows x 8 neighbors

    const int tid  = threadIdx.x;
    const int lane = tid & 31;
    const int wid  = tid >> 5;

    // Stage W2 hi/lo fragments (issued early; overlaps with phase 1 latency).
    for (int i = tid; i < 8192; i += 256) { sH[i] = W2h[i]; sL[i] = W2l[i]; }

    // ------------------------- Phase 1: KNN -------------------------------
    for (int rr = 0; rr < 2; ++rr) {
        const int b_loc = wid * 2 + rr;
        const int b     = blockIdx.x * 16 + b_loc;

        const float2 g = ((const float2*)Gl)[b];

        float vals[8]; int idxs[8];
#pragma unroll
        for (int j = 0; j < 8; ++j) { vals[j] = 3.4e38f; idxs[j] = 0; }

        const float4* a4   = (const float4*)ancL;
        const size_t base4 = ((size_t)b * MANC) >> 1;
        for (int m0 = 0; m0 < MANC; m0 += 128) {
            size_t p  = base4 + (size_t)(m0 >> 1) + (size_t)lane * 2;
            float4 q0 = a4[p];
            float4 q1 = a4[p + 1];
            int mb = m0 + lane * 4;
            float xs[4] = {q0.x, q0.z, q1.x, q1.z};
            float ys[4] = {q0.y, q0.w, q1.y, q1.w};
#pragma unroll
            for (int t = 0; t < 4; ++t) {
                float dx = xs[t] - g.x, dy = ys[t] - g.y;
                float d  = dx * dx + dy * dy;
                if (d < vals[7]) {
                    vals[7] = d; idxs[7] = mb + t;
#pragma unroll
                    for (int j = 7; j > 0; --j) {
                        if (vals[j] < vals[j - 1]) {
                            float tv = vals[j]; vals[j] = vals[j - 1]; vals[j - 1] = tv;
                            int   ti = idxs[j]; idxs[j] = idxs[j - 1]; idxs[j - 1] = ti;
                        }
                    }
                }
            }
        }

        // Cross-lane merge: 8 rounds of wave-argmin over per-lane sorted heads.
        float sel_d[8]; int sel_i[8];
#pragma unroll
        for (int r = 0; r < 8; ++r) {
            float m = vals[0];
#pragma unroll
            for (int off = 16; off >= 1; off >>= 1) m = fminf(m, __shfl_xor(m, off, 32));
            unsigned long long bal = __ballot(vals[0] == m);
            int winner = __ffsll(bal) - 1;
            sel_d[r] = m;
            sel_i[r] = __shfl(idxs[0], winner, 32);
            bool isw = (lane == winner);
#pragma unroll
            for (int j = 0; j < 7; ++j) {
                vals[j] = isw ? vals[j + 1] : vals[j];
                idxs[j] = isw ? idxs[j + 1] : idxs[j];
            }
            vals[7] = isw ? 3.4e38f : vals[7];
        }

        // softmax(d2 / tau) with max subtraction (sel_d ascending)
        float mx = sel_d[7];
        float w[8]; float s = 0.f;
#pragma unroll
        for (int r = 0; r < 8; ++r) { w[r] = __expf((sel_d[r] - mx) * (1.0f / TAU)); s += w[r]; }
        float inv = 1.0f / s;

        if (lane == 0) {
            const float2* a2 = (const float2*)ancL;
#pragma unroll
            for (int r = 0; r < 8; ++r) {
                float2 c = a2[(size_t)b * MANC + sel_i[r]];
                smeta[b_loc * KSEL + r] = make_float4(c.x, c.y, w[r] * inv, sel_d[r]);
            }
        }
    }
    __syncthreads();

    // ------------------------- Phase 2: WMMA MLP --------------------------
    const int hi16 = lane >> 4;                 // 0 for lanes 0-15, 1 for 16-31
    const int lrow = wid * 16 + (lane & 15);    // local GEMM row owned for A

    const float4 mt = smeta[lrow];
    const float ax = mt.x, ay = mt.y;

    // Build A fragments: h1 = gelu(W1 @ [x,y] + b1), split to bf16 hi/lo.
    // ISA 16-bit A 16x32 layout: lane<16 VGPR j holds K = 2j (+8 if j>=4);
    // lanes 16-31 add +8.
    union AFrag { v16bf v; __bf16 e[16]; };
    AFrag Ah[4], Al[4];
    const int kbaseA = hi16 ? 8 : 0;
#pragma unroll
    for (int c = 0; c < 4; ++c) {
#pragma unroll
        for (int j = 0; j < 8; ++j) {
            const int k0 = 2 * j + ((j >= 4) ? 8 : 0) + kbaseA;
#pragma unroll
            for (int h = 0; h < 2; ++h) {
                const int e = c * 32 + k0 + h;
                float hv = ax * W1[2 * e] + ay * W1[2 * e + 1] + b1[e];
                hv = gelu_exact(hv);
                __bf16 hb = (__bf16)hv;
                Ah[c].e[2 * j + h] = hb;
                Al[c].e[2 * j + h] = (__bf16)(hv - (float)hb);
            }
        }
    }

    union F8 { v8f v; float e[8]; };
    F8 acc[8];
#pragma unroll
    for (int t = 0; t < 8; ++t)
#pragma unroll
        for (int r = 0; r < 8; ++r) acc[t].e[r] = 0.f;

    // B fragments from LDS: B[k][n] = W2[f=n][e=k]; VGPR j holds e-pair
    // ep = c*16 + j (+4 if j>=4) (+4 for lanes 16-31), column f = 16*t + lane%16.
    const int n      = lane & 15;
    const int kbaseB = hi16 ? 4 : 0;   // in e-pair units
#pragma unroll
    for (int t = 0; t < 8; ++t) {
        const int f = 16 * t + n;
#pragma unroll
        for (int c = 0; c < 4; ++c) {
            union { v16bf v; unsigned u[8]; } Bh, Bl;
#pragma unroll
            for (int j = 0; j < 8; ++j) {
                const int ep  = c * 16 + j + ((j >= 4) ? 4 : 0) + kbaseB;
                const int idx = ep * 128 + ((f + ep) & 127);
                Bh.u[j] = sH[idx];
                Bl.u[j] = sL[idx];
            }
            acc[t].v = __builtin_amdgcn_wmma_f32_16x16x32_bf16(false, Ah[c].v, false, Bh.v,
                                                               (short)0, acc[t].v, false, false);
            acc[t].v = __builtin_amdgcn_wmma_f32_16x16x32_bf16(false, Ah[c].v, false, Bl.v,
                                                               (short)0, acc[t].v, false, false);
            acc[t].v = __builtin_amdgcn_wmma_f32_16x16x32_bf16(false, Al[c].v, false, Bh.v,
                                                               (short)0, acc[t].v, false, false);
        }
    }

    // C layout: VGPR r = rows M=r (lanes 0-15) / M=r+8 (lanes 16-31).
    // Lanes 0-15 reduce batch row (wid*2), lanes 16-31 reduce (wid*2+1).
    const int bloc_out = wid * 2 + hi16;
    const float* smetaF = (const float*)smeta;
    float wk[8];
#pragma unroll
    for (int r = 0; r < 8; ++r) wk[r] = smetaF[(bloc_out * 8 + r) * 4 + 2];

    const size_t bglob = (size_t)blockIdx.x * 16 + bloc_out;
#pragma unroll
    for (int t = 0; t < 8; ++t) {
        const int f = 16 * t + n;
        const float bias = b2[f];
        float o = 0.f;
#pragma unroll
        for (int r = 0; r < 8; ++r) o += wk[r] * gelu_exact(acc[t].e[r] + bias);
        out[bglob * EMB + f] = o;
    }
}

// ---------------------------------------------------------------------------
extern "C" void kernel_launch(void* const* d_in, const int* in_sizes, int n_in,
                              void* d_out, int out_size, void* d_ws, size_t ws_size,
                              hipStream_t stream) {
    const float* Gl   = (const float*)d_in[0];
    const float* ancL = (const float*)d_in[1];
    const float* W1   = (const float*)d_in[2];
    const float* b1   = (const float*)d_in[3];
    const float* W2   = (const float*)d_in[4];
    const float* b2   = (const float*)d_in[5];
    float* out = (float*)d_out;

    unsigned* W2h = (unsigned*)d_ws;                     // 32 KB
    unsigned* W2l = (unsigned*)((char*)d_ws + 32768);    // 32 KB

    k_split_w2<<<32, 256, 0, stream>>>(W2, W2h, W2l);
    k_fused<<<BATCH / 16, 256, 0, stream>>>(Gl, ancL, W1, b1, b2, W2h, W2l, out);
}
